// BiRNNGenerator_89558658056462
// MI455X (gfx1250) — compile-verified, matrix operand order checked by reading, not run on hardware
//
#include <hip/hip_runtime.h>

// Problem dims (from reference)
#define S_LEN 128
#define T_LEN 64
#define BSZ   64
#define EMBD  256
#define HIDD  512
#define DHID  1024
#define VOCS  32000

typedef __bf16 bf16;
typedef __attribute__((ext_vector_type(16))) __bf16 v16bf;
typedef __attribute__((ext_vector_type(8)))  __bf16 v8bf;
typedef __attribute__((ext_vector_type(8)))  float  v8f;

// ---------------- device helpers ----------------

__device__ __forceinline__ unsigned short f2bf_raw(float f) {
  union { float f; unsigned u; } v; v.f = f;
  unsigned r = v.u + 0x7FFFu + ((v.u >> 16) & 1u);   // round-nearest-even
  return (unsigned short)(r >> 16);
}

__device__ __forceinline__ float sigf(float x) { return 1.0f / (1.0f + __expf(-x)); }

// 16-bit A/B fragment for v_wmma_f32_16x16x32_bf16 (wave32):
// lane L holds row m = L&15; khalf = L>>4.
// elements 0..7  <- K = 8*khalf + (0..7)        (contiguous 16B)
// elements 8..15 <- K = 16 + 8*khalf + (0..7)   (contiguous 16B)
__device__ __forceinline__ v16bf load_frag(const bf16* __restrict__ rowp, int khalf) {
  v8bf lo = *reinterpret_cast<const v8bf*>(rowp + 8 * khalf);
  v8bf hi = *reinterpret_cast<const v8bf*>(rowp + 16 + 8 * khalf);
  v16bf r;
#pragma unroll
  for (int i = 0; i < 8; ++i) { r[i] = lo[i]; r[i + 8] = hi[i]; }
  return r;
}

// ---------------- WMMA GEMM: C(MxN) = A(MxK) * W(NxK)^T + addend + bias ----------------
// A, W bf16 row-major; C fp32. addend (optional) fp32 MxN (ld=lad), bias (optional) fp32 N.
// Block = 256 threads = 8 waves; wave computes a 32x64 strip (2 M-tiles x 4 N-tiles):
// each B fragment feeds two WMMAs -> ~22 FLOP/byte, halving L2 B-traffic vs 16x64.
// grid = (ceil(N/512), M/32).  Requires: M%32==0, N%64==0, K%32==0.

__global__ void __launch_bounds__(256)
wmma_gemm_bf16(const unsigned short* __restrict__ Au, int lda,
               const unsigned short* __restrict__ Wu, int ldb,
               float* __restrict__ C, int ldc,
               const float* __restrict__ addend, int lad,
               const float* __restrict__ bias,
               int N, int K) {
  const bf16* A = reinterpret_cast<const bf16*>(Au);
  const bf16* W = reinterpret_cast<const bf16*>(Wu);
  const int wave  = threadIdx.x >> 5;
  const int lane  = threadIdx.x & 31;
  const int khalf = lane >> 4;
  const int lidx  = lane & 15;
  const int m0      = blockIdx.y << 5;
  const int colBase = (blockIdx.x * 8 + wave) << 6;
  if (colBase >= N) return;                 // wave-uniform: EXEC stays all-ones

  const bf16* arow0 = A + (long)(m0 + lidx) * lda;
  const bf16* arow1 = A + (long)(m0 + 16 + lidx) * lda;
  const bf16* wr0   = W + (long)(colBase +  0 + lidx) * ldb;
  const bf16* wr1   = W + (long)(colBase + 16 + lidx) * ldb;
  const bf16* wr2   = W + (long)(colBase + 32 + lidx) * ldb;
  const bf16* wr3   = W + (long)(colBase + 48 + lidx) * ldb;

  v8f acc00 = {}, acc01 = {}, acc02 = {}, acc03 = {};
  v8f acc10 = {}, acc11 = {}, acc12 = {}, acc13 = {};
  for (int k0 = 0; k0 < K; k0 += 32) {
    if (k0 + 32 < K) {
      __builtin_prefetch(arow0 + k0 + 32, 0, 1);
      __builtin_prefetch(wr0 + k0 + 32, 0, 1);
      __builtin_prefetch(wr2 + k0 + 32, 0, 1);
    }
    // issue all fragment loads first, then 8 independent WMMAs
    v16bf a0 = load_frag(arow0 + k0, khalf);
    v16bf a1 = load_frag(arow1 + k0, khalf);
    v16bf b0 = load_frag(wr0 + k0, khalf);
    v16bf b1 = load_frag(wr1 + k0, khalf);
    v16bf b2 = load_frag(wr2 + k0, khalf);
    v16bf b3 = load_frag(wr3 + k0, khalf);
    acc00 = __builtin_amdgcn_wmma_f32_16x16x32_bf16(false, a0, false, b0, (short)0, acc00, false, false);
    acc10 = __builtin_amdgcn_wmma_f32_16x16x32_bf16(false, a1, false, b0, (short)0, acc10, false, false);
    acc01 = __builtin_amdgcn_wmma_f32_16x16x32_bf16(false, a0, false, b1, (short)0, acc01, false, false);
    acc11 = __builtin_amdgcn_wmma_f32_16x16x32_bf16(false, a1, false, b1, (short)0, acc11, false, false);
    acc02 = __builtin_amdgcn_wmma_f32_16x16x32_bf16(false, a0, false, b2, (short)0, acc02, false, false);
    acc12 = __builtin_amdgcn_wmma_f32_16x16x32_bf16(false, a1, false, b2, (short)0, acc12, false, false);
    acc03 = __builtin_amdgcn_wmma_f32_16x16x32_bf16(false, a0, false, b3, (short)0, acc03, false, false);
    acc13 = __builtin_amdgcn_wmma_f32_16x16x32_bf16(false, a1, false, b3, (short)0, acc13, false, false);
  }

  v8f accs[2][4] = {{acc00, acc01, acc02, acc03}, {acc10, acc11, acc12, acc13}};
#pragma unroll
  for (int mt = 0; mt < 2; ++mt) {
#pragma unroll
    for (int t = 0; t < 4; ++t) {
      const int n = colBase + t * 16 + lidx;
      const float bz = bias ? bias[n] : 0.0f;
#pragma unroll
      for (int r = 0; r < 8; ++r) {
        const int m = m0 + mt * 16 + r + 8 * khalf;  // C/D: VGPR r <-> row +r+8*khalf
        float val = accs[mt][t][r] + bz;
        if (addend) val += addend[(long)m * lad + n];
        C[(long)m * ldc + n] = val;
      }
    }
  }
}

// ---------------- LSTM gates (i,f,g,o order) ----------------
__global__ void __launch_bounds__(256)
lstm_gate(const float* __restrict__ g, int H,
          float* __restrict__ c, int ldcell,
          unsigned short* __restrict__ hbf, int ldh,
          unsigned short* __restrict__ ybf, int ldy,
          int B) {
  int idx = blockIdx.x * 256 + threadIdx.x;
  if (idx >= B * H) return;
  int b = idx / H, j = idx - b * H;
  const float* gr = g + (long)b * 4 * H;
  float gi = gr[j], gf = gr[H + j], gg = gr[2 * H + j], go = gr[3 * H + j];
  float cn = sigf(gf) * c[(long)b * ldcell + j] + sigf(gi) * tanhf(gg);
  float hn = sigf(go) * tanhf(cn);
  c[(long)b * ldcell + j] = cn;
  unsigned short hb = f2bf_raw(hn);
  hbf[(long)b * ldh + j] = hb;
  if (ybf) ybf[(long)b * ldy + j] = hb;
}

// ---------------- conversion / embedding / init ----------------
__global__ void __launch_bounds__(256)
cvt_f32_bf16(const float* __restrict__ in, unsigned short* __restrict__ out, long n) {
  long i = (long)blockIdx.x * 256 + threadIdx.x;
  long stride = (long)gridDim.x * 256;
  for (; i < n; i += stride) out[i] = f2bf_raw(in[i]);
}

__global__ void __launch_bounds__(256)
embed_gather_bf16(const int* __restrict__ tok, const float* __restrict__ emb,
                  unsigned short* __restrict__ out, long n) {
  long i = (long)blockIdx.x * 256 + threadIdx.x;
  if (i >= n) return;
  int t = (int)(i >> 8);       // EMBD == 256
  int e = (int)(i & 255);
  out[i] = f2bf_raw(emb[(long)tok[t] * EMBD + e]);
}

__global__ void __launch_bounds__(256)
embed_dec_bf16(const int* __restrict__ sent, const int* __restrict__ tgt,
               const float* __restrict__ emb, unsigned short* __restrict__ out) {
  long i = (long)blockIdx.x * 256 + threadIdx.x;
  if (i >= (long)T_LEN * BSZ * EMBD) return;
  int e  = (int)(i & 255);
  int tb = (int)(i >> 8);
  int t  = tb / BSZ, b = tb - t * BSZ;
  int tok = (t == 0) ? sent[(S_LEN - 1) * BSZ + b] : tgt[(t - 1) * BSZ + b];
  out[i] = f2bf_raw(emb[(long)tok * EMBD + e]);
}

__global__ void __launch_bounds__(256)
zero_f32_slice(float* __restrict__ p, int rows, int cols, int ld) {
  int i = blockIdx.x * 256 + threadIdx.x;
  if (i >= rows * cols) return;
  int r = i / cols, cc = i - r * cols;
  p[(long)r * ld + cc] = 0.0f;
}

__global__ void __launch_bounds__(256)
zero_bf16_slice(unsigned short* __restrict__ p, int rows, int cols, int ld) {
  int i = blockIdx.x * 256 + threadIdx.x;
  if (i >= rows * cols) return;
  int r = i / cols, cc = i - r * cols;
  p[(long)r * ld + cc] = 0;
}

// ---------------- in-place row log-softmax (online max/sum) ----------------
__global__ void __launch_bounds__(256)
logsoftmax_rows(float* __restrict__ x, int ncols) {
  __shared__ float sm[256], ss[256];
  float* p = x + (long)blockIdx.x * ncols;
  int tid = threadIdx.x;
  float m = -3.4e38f, s = 0.0f;
  for (int c = tid; c < ncols; c += 256) {
    float v = p[c];
    if (v > m) { s = s * __expf(m - v) + 1.0f; m = v; }
    else       { s += __expf(v - m); }
  }
  sm[tid] = m; ss[tid] = s;
  __syncthreads();
  for (int off = 128; off > 0; off >>= 1) {
    if (tid < off) {
      float m2 = sm[tid + off], s2 = ss[tid + off];
      float mm = fmaxf(sm[tid], m2);
      ss[tid] = ss[tid] * __expf(sm[tid] - mm) + s2 * __expf(m2 - mm);
      sm[tid] = mm;
    }
    __syncthreads();
  }
  float M = sm[0], L = __logf(ss[0]);
  for (int c = tid; c < ncols; c += 256) p[c] = p[c] - M - L;
}

// ---------------- host orchestration ----------------

extern "C" void kernel_launch(void* const* d_in, const int* in_sizes, int n_in,
                              void* d_out, int out_size, void* d_ws, size_t ws_size,
                              hipStream_t stream) {
  (void)in_sizes; (void)n_in; (void)out_size; (void)ws_size;
  const int*   sentences = (const int*)  d_in[0];
  const int*   targets   = (const int*)  d_in[1];
  const float* emb       = (const float*)d_in[2];
  const float* e0_Wih = (const float*)d_in[3],  *e0_Whh = (const float*)d_in[4];
  const float* e0_bih = (const float*)d_in[5],  *e0_bhh = (const float*)d_in[6];
  const float* e1_Wih = (const float*)d_in[7],  *e1_Whh = (const float*)d_in[8];
  const float* e1_bih = (const float*)d_in[9],  *e1_bhh = (const float*)d_in[10];
  const float* d0_Wih = (const float*)d_in[11], *d0_Whh = (const float*)d_in[12];
  const float* d0_bih = (const float*)d_in[13], *d0_bhh = (const float*)d_in[14];
  const float* d1_Wih = (const float*)d_in[15], *d1_Whh = (const float*)d_in[16];
  const float* d1_bih = (const float*)d_in[17], *d1_bhh = (const float*)d_in[18];
  const float* W_out  = (const float*)d_in[19], *b_out  = (const float*)d_in[20];

  const int G4H = 4 * HIDD;   // 2048
  const int G4D = 4 * DHID;   // 4096

  // workspace carve-out (256B aligned)
  char* wsb = (char*)d_ws; size_t off = 0;
  auto alloc = [&](size_t bytes) -> char* {
    char* p = wsb + off; off = (off + bytes + 255) & ~(size_t)255; return p;
  };
  typedef unsigned short us;
  us* w_e0_Wih = (us*)alloc((size_t)2 * G4H * EMBD * 2);
  us* w_e0_Whh = (us*)alloc((size_t)2 * G4H * HIDD * 2);
  us* w_e1_Wih = (us*)alloc((size_t)2 * G4H * DHID * 2);
  us* w_e1_Whh = (us*)alloc((size_t)2 * G4H * HIDD * 2);
  us* w_d0_Wih = (us*)alloc((size_t)G4D * EMBD * 2);
  us* w_d0_Whh = (us*)alloc((size_t)G4D * DHID * 2);
  us* w_d1_Wih = (us*)alloc((size_t)G4D * DHID * 2);
  us* w_d1_Whh = (us*)alloc((size_t)G4D * DHID * 2);
  us* w_out_bf = (us*)alloc((size_t)VOCS * DHID * 2);
  us* x_bf   = (us*)alloc((size_t)S_LEN * BSZ * EMBD * 2);
  us* y0_bf  = (us*)alloc((size_t)S_LEN * BSZ * DHID * 2);   // enc layer0 output [yf|yb]
  us* xd_bf  = (us*)alloc((size_t)T_LEN * BSZ * EMBD * 2);
  us* yd0_bf = (us*)alloc((size_t)T_LEN * BSZ * DHID * 2);
  us* yd1_bf = (us*)alloc((size_t)T_LEN * BSZ * DHID * 2);
  us* hbf0   = (us*)alloc((size_t)BSZ * DHID * 2);           // layer0 state h (bf16)
  us* hbf1   = (us*)alloc((size_t)BSZ * DHID * 2);           // layer1 state h
  float* xwbuf = (float*)alloc((size_t)S_LEN * BSZ * G4H * 4);  // 64MB, reused everywhere
  float* gbuf  = (float*)alloc((size_t)BSZ * G4D * 4);
  float* c0    = (float*)alloc((size_t)BSZ * DHID * 4);
  float* c1    = (float*)alloc((size_t)BSZ * DHID * 4);

  auto cvt = [&](const float* src, us* dst, long n) {
    int blocks = (int)((n + 255) / 256); if (blocks > 16384) blocks = 16384;
    cvt_f32_bf16<<<blocks, 256, 0, stream>>>(src, dst, n);
  };
  auto gemm = [&](const us* A, int lda, const us* W, int ldb, float* C, int ldc,
                  const float* addend, int lad, const float* bias, int M, int N, int K) {
    dim3 grid((unsigned)((N + 511) / 512), (unsigned)(M / 32));
    wmma_gemm_bf16<<<grid, 256, 0, stream>>>(A, lda, W, ldb, C, ldc, addend, lad, bias, N, K);
  };

  // 1) weights -> bf16
  cvt(e0_Wih, w_e0_Wih, (long)2 * G4H * EMBD);
  cvt(e0_Whh, w_e0_Whh, (long)2 * G4H * HIDD);
  cvt(e1_Wih, w_e1_Wih, (long)2 * G4H * DHID);
  cvt(e1_Whh, w_e1_Whh, (long)2 * G4H * HIDD);
  cvt(d0_Wih, w_d0_Wih, (long)G4D * EMBD);
  cvt(d0_Whh, w_d0_Whh, (long)G4D * DHID);
  cvt(d1_Wih, w_d1_Wih, (long)G4D * DHID);
  cvt(d1_Whh, w_d1_Whh, (long)G4D * DHID);
  cvt(W_out,  w_out_bf, (long)VOCS * DHID);

  // 2) embeddings
  {
    long n = (long)S_LEN * BSZ * EMBD;
    embed_gather_bf16<<<(unsigned)((n + 255) / 256), 256, 0, stream>>>(sentences, emb, x_bf, n);
    long nd = (long)T_LEN * BSZ * EMBD;
    embed_dec_bf16<<<(unsigned)((nd + 255) / 256), 256, 0, stream>>>(sentences, targets, emb, xd_bf);
  }

  // 3) encoder: 2 layers x 2 directions, sequential
  for (int l = 0; l < 2; ++l) {
    const us*   Wih  = (l == 0) ? w_e0_Wih : w_e1_Wih;
    const us*   Whh  = (l == 0) ? w_e0_Whh : w_e1_Whh;
    const float* bih = (l == 0) ? e0_bih   : e1_bih;
    const float* bhh = (l == 0) ? e0_bhh   : e1_bhh;
    const us*   Xin  = (l == 0) ? x_bf     : y0_bf;
    const int   Kin  = (l == 0) ? EMBD     : DHID;
    us*   hbf = (l == 0) ? hbf0 : hbf1;
    float* cc = (l == 0) ? c0   : c1;
    for (int d = 0; d < 2; ++d) {
      // xw = X @ Wih[d]^T + bih[d]   (S*B x 4H)
      gemm(Xin, Kin, Wih + (size_t)d * G4H * Kin, Kin, xwbuf, G4H,
           nullptr, 0, bih + d * G4H, S_LEN * BSZ, G4H, Kin);
      // zero h, c slices for this direction
      int ne = BSZ * HIDD;
      zero_f32_slice<<<(ne + 255) / 256, 256, 0, stream>>>(cc + d * HIDD, BSZ, HIDD, DHID);
      zero_bf16_slice<<<(ne + 255) / 256, 256, 0, stream>>>(hbf + d * HIDD, BSZ, HIDD, DHID);
      // recurrent scan
      for (int t = 0; t < S_LEN; ++t) {
        int tt = d ? (S_LEN - 1 - t) : t;
        gemm(hbf + d * HIDD, DHID, Whh + (size_t)d * G4H * HIDD, HIDD, gbuf, G4H,
             xwbuf + (size_t)tt * BSZ * G4H, G4H, bhh + d * G4H, BSZ, G4H, HIDD);
        us* ybf = (l == 0) ? (y0_bf + (size_t)tt * BSZ * DHID + d * HIDD) : nullptr;
        lstm_gate<<<(BSZ * HIDD + 255) / 256, 256, 0, stream>>>(
            gbuf, HIDD, cc + d * HIDD, DHID, hbf + d * HIDD, DHID, ybf, DHID, BSZ);
      }
    }
  }
  // encoder final states (hbf0/c0, hbf1/c1) == decoder initial states (B x 1024)

  // 4) decoder: 2 layers, unidirectional
  for (int l = 0; l < 2; ++l) {
    const us*   Wih  = (l == 0) ? w_d0_Wih : w_d1_Wih;
    const us*   Whh  = (l == 0) ? w_d0_Whh : w_d1_Whh;
    const float* bih = (l == 0) ? d0_bih   : d1_bih;
    const float* bhh = (l == 0) ? d0_bhh   : d1_bhh;
    const us*   Xin  = (l == 0) ? xd_bf    : yd0_bf;
    const int   Kin  = (l == 0) ? EMBD     : DHID;
    us*   ybuf = (l == 0) ? yd0_bf : yd1_bf;
    us*   hbf  = (l == 0) ? hbf0 : hbf1;
    float* cc  = (l == 0) ? c0   : c1;
    gemm(Xin, Kin, Wih, Kin, xwbuf, G4D, nullptr, 0, bih, T_LEN * BSZ, G4D, Kin);
    for (int t = 0; t < T_LEN; ++t) {
      gemm(hbf, DHID, Whh, DHID, gbuf, G4D,
           xwbuf + (size_t)t * BSZ * G4D, G4D, bhh, BSZ, G4D, DHID);
      lstm_gate<<<(BSZ * DHID + 255) / 256, 256, 0, stream>>>(
          gbuf, DHID, cc, DHID, hbf, DHID, ybuf + (size_t)t * BSZ * DHID, DHID, BSZ);
    }
  }

  // 5) logits = yd1 @ W_out^T + b_out -> d_out (fp32), then in-place log-softmax
  float* logits = (float*)d_out;
  gemm(yd1_bf, DHID, w_out_bf, DHID, logits, VOCS, nullptr, 0, b_out,
       T_LEN * BSZ, VOCS, DHID);
  logsoftmax_rows<<<T_LEN * BSZ, 256, 0, stream>>>(logits, VOCS);
}
